// LSTMBase_12833362281122
// MI455X (gfx1250) — matrix operand order repaired and checked by reading, not compiled, not run on hardware
//
#include <hip/hip_runtime.h>
#include <hip/hip_bf16.h>

// LSTM: B=64, T=2048, I=512, H=512 (4H = 2048)
//
// Phase 1: x_gates = inputs @ W_i + bias_i + bias_h   (bf16 WMMA, f32 acc)
// Phase 2: persistent 32-WG scan; per step:
//   - TDM (tensor_load_to_lds) streams next step's x_gates tiles into LDS,
//     double buffered, overlapping the recurrent WMMA chain (TENSORcnt sync)
//   - g = h @ W_h via WMMA (W_h fragments VGPR-resident for all 2048 steps)
//   - gate nonlinearity + cell update WG-locally in LDS
//   - h broadcast via global bf16 double buffer; 32-arrival atomic barrier

typedef __attribute__((ext_vector_type(16))) __bf16 v16bf;
typedef __attribute__((ext_vector_type(8)))  float  v8f;
typedef unsigned int u32x4 __attribute__((ext_vector_type(4)));
typedef unsigned int u32x8 __attribute__((ext_vector_type(8)));

#define B_   64
#define T_   2048
#define I_   512
#define H_   512
#define G4_  2048          // 4*H
#define KT_  16            // K/32 for both K=512 GEMMs
#define M_X  (B_ * T_)     // 131072 rows in phase-1 GEMM
#define NWG_SCAN 32

// ---------------------------------------------------------------------------
// Fragment helpers (ISA 7.12.2 layouts, stored lane-major: each lane's 16
// halves are 32B contiguous).
// ---------------------------------------------------------------------------
__device__ __forceinline__ size_t frag_off(int t0, int kt, int lane) {
  return ((size_t)(t0 * KT_ + kt) * 32 + (size_t)lane) * 16;
}

__device__ __forceinline__ v8f wmma_bf16(v16bf a, v16bf b, v8f c) {
  return __builtin_amdgcn_wmma_f32_16x16x32_bf16(
      false, a, false, b, (short)0, c, false, false);
}

__device__ __forceinline__ float sigf(float x) {
  return 1.0f / (1.0f + __expf(-x));
}

// ---------------------------------------------------------------------------
// TDM: issue a 2D tensor_load_to_lds of a 16(cols) x 64(rows) f32 tile.
//   global rows strided by T*4H floats, written back-to-back into LDS.
// D# per CDNA5 ISA (groups 2/3 zero: 2D tile).
// ---------------------------------------------------------------------------
__device__ __forceinline__ void tdm_load_xg_tile(unsigned long long gaddr,
                                                 unsigned lds_addr) {
  u32x4 g0;
  g0.x = 1u;                                  // count=1 (valid descriptor)
  g0.y = lds_addr;                            // LDS byte address
  g0.z = (unsigned)gaddr;                     // global addr lo
  g0.w = (unsigned)((gaddr >> 32) & 0x01FFFFFFu) | 0x80000000u;  // hi | type=2
  u32x8 g1;
  g1[0] = 2u << 16;                           // data_size=4B; no mask/pad
  g1[1] = 16u << 16;                          // tensor_dim0 = 16
  g1[2] = 64u << 16;                          // tensor_dim1 = 64
  g1[3] = 16u << 16;                          // tile_dim0 = 16
  g1[4] = 64u;                                // tile_dim1 = 64 (tile_dim2=0)
  g1[5] = (unsigned)((size_t)T_ * G4_);       // tensor_dim0_stride lo (4M)
  g1[6] = 0u;                                 // stride hi / dim1_stride lo
  g1[7] = 0u;
  u32x4 gz = {0u, 0u, 0u, 0u};                // groups 2/3: unused (2D)
  asm volatile("tensor_load_to_lds %0, %1, %2, %3"
               :: "s"(g0), "s"(g1), "s"(gz), "s"(gz)
               : "memory");
}

// ---------------------------------------------------------------------------
// Init: zero h double-buffer + barrier counter (runs every launch/replay).
// ---------------------------------------------------------------------------
__global__ void lstm_init_kernel(unsigned* __restrict__ counter,
                                 __bf16* __restrict__ hbuf, int nhalf) {
  int gid = blockIdx.x * blockDim.x + threadIdx.x;
  if (gid == 0) *counter = 0u;
  for (int i = gid; i < nhalf; i += gridDim.x * blockDim.x)
    hbuf[i] = (__bf16)0.0f;
}

// ---------------------------------------------------------------------------
// Pack row-major f32 X[M][ldx] into bf16 A-fragment layout.
// ---------------------------------------------------------------------------
__global__ void pack_a_kernel(const float* __restrict__ X,
                              __bf16* __restrict__ P,
                              int mtiles, int ldx) {
  int gid = blockIdx.x * blockDim.x + threadIdx.x;
  int total = mtiles * KT_ * 32;
  if (gid >= total) return;
  int lane = gid & 31;
  int tile = gid >> 5;            // tile = mt*KT_ + kt
  int kt = tile & (KT_ - 1);
  int mt = tile / KT_;
  int m = mt * 16 + (lane & 15);
  int kbase = kt * 32 + ((lane >> 4) & 1) * 8;
  const float* src = X + (size_t)m * ldx + kbase;
  __bf16* dst = P + (size_t)gid * 16;
#pragma unroll
  for (int j = 0; j < 8; ++j) dst[j] = (__bf16)src[j];
#pragma unroll
  for (int j = 0; j < 8; ++j) dst[8 + j] = (__bf16)src[16 + j];
}

// ---------------------------------------------------------------------------
// Pack row-major f32 W[K][ldw] into bf16 B-fragment layout.
// ---------------------------------------------------------------------------
__global__ void pack_b_kernel(const float* __restrict__ W,
                              __bf16* __restrict__ P,
                              int ntiles, int ldw) {
  int gid = blockIdx.x * blockDim.x + threadIdx.x;
  int total = ntiles * KT_ * 32;
  if (gid >= total) return;
  int lane = gid & 31;
  int tile = gid >> 5;            // tile = nt*KT_ + kt
  int kt = tile & (KT_ - 1);
  int nt = tile / KT_;
  int n = nt * 16 + (lane & 15);
  int kstart = kt * 32 + ((lane >> 4) & 1) * 8;
  __bf16* dst = P + (size_t)gid * 16;
#pragma unroll
  for (int p = 0; p < 16; ++p)
    dst[p] = (__bf16)W[(size_t)(kstart + p + (p >= 8 ? 8 : 0)) * ldw + n];
}

// ---------------------------------------------------------------------------
// Phase 1: xg[M_X][2048] = packed(inputs) @ packed(W_i) + bias_i + bias_h
// ---------------------------------------------------------------------------
__global__ __launch_bounds__(256) void gemm_xgates_kernel(
    const __bf16* __restrict__ A, const __bf16* __restrict__ Bp,
    const float* __restrict__ bias_i, const float* __restrict__ bias_h,
    float* __restrict__ xg) {
  int blk = blockIdx.x;
  int bm = blk & 2047;        // 131072/64 m-blocks
  int bn = blk >> 11;         // 2048/128  n-blocks
  int w = threadIdx.x >> 5;
  int lane = threadIdx.x & 31;
  int mtB = bm * 4 + (w & 1) * 2;
  int ntB = bn * 8 + (w >> 1) * 2;

  v8f acc00 = {}, acc01 = {}, acc10 = {}, acc11 = {};
#pragma unroll 4
  for (int kt = 0; kt < KT_; ++kt) {
    v16bf a0 = *(const v16bf*)(A + frag_off(mtB,     kt, lane));
    v16bf a1 = *(const v16bf*)(A + frag_off(mtB + 1, kt, lane));
    v16bf b0 = *(const v16bf*)(Bp + frag_off(ntB,     kt, lane));
    v16bf b1 = *(const v16bf*)(Bp + frag_off(ntB + 1, kt, lane));
    acc00 = wmma_bf16(a0, b0, acc00);
    acc01 = wmma_bf16(a0, b1, acc01);
    acc10 = wmma_bf16(a1, b0, acc10);
    acc11 = wmma_bf16(a1, b1, acc11);
  }

  int nn = lane & 15;
  int rbase = (lane >> 4) ? 8 : 0;
  int n0 = ntB * 16 + nn;
  int n1 = (ntB + 1) * 16 + nn;
  float bs0 = bias_i[n0] + bias_h[n0];
  float bs1 = bias_i[n1] + bias_h[n1];
#pragma unroll
  for (int r = 0; r < 8; ++r) {
    int m0 = mtB * 16 + rbase + r;
    int m1 = (mtB + 1) * 16 + rbase + r;
    xg[(size_t)m0 * G4_ + n0] = acc00[r] + bs0;
    xg[(size_t)m0 * G4_ + n1] = acc01[r] + bs1;
    xg[(size_t)m1 * G4_ + n0] = acc10[r] + bs0;
    xg[(size_t)m1 * G4_ + n1] = acc11[r] + bs1;
  }
}

// ---------------------------------------------------------------------------
// Phase 2: persistent scan. WG `wgid` owns hidden columns [wgid*16, +16)
// across all four gates. W_h fragments VGPR-resident; x_gates tiles arrive
// via TDM into a double-buffered LDS staging area one step ahead.
// ---------------------------------------------------------------------------
__global__ __launch_bounds__(256) void lstm_scan_kernel(
    const float* __restrict__ xg, const __bf16* __restrict__ whp,
    __bf16* __restrict__ hbuf0, __bf16* __restrict__ hbuf1,
    float* __restrict__ out, unsigned* __restrict__ counter) {
  __shared__ float gx[64 * 64];          // recurrent-GEMM exchange (16 KB)
  __shared__ float cst[64 * 16];         // persistent cell state (4 KB)
  __shared__ float xgl[2][4][64][16];    // TDM-staged x_gates tiles (32 KB)

  const int wgid = blockIdx.x;           // 0..31 -> hidden column group
  const int tid = threadIdx.x;
  const int w = tid >> 5;
  const int lane = tid & 31;
  const int gate = w >> 1;               // 0..3 (GEMM gate of this wave)
  const int mtb = (w & 1) * 2;           // m-tile pair base
  const int nn = lane & 15;
  const int hi = (lane >> 4) & 1;
  const int nt0 = gate * 32 + wgid;      // B n-tile index within 4H

  for (int e = tid; e < 64 * 16; e += 256) cst[e] = 0.0f;
  __syncthreads();

  // W_h fragments: loop-invariant -> VGPR resident for the whole scan.
  v16bf bfrag[KT_];
#pragma unroll
  for (int kt = 0; kt < KT_; ++kt)
    bfrag[kt] = *(const v16bf*)(whp + frag_off(nt0, kt, lane));

  // TDM setup: waves 0..3 each own one gate's x_gates tile stream.
  const int tg = w;                      // TDM gate id for waves 0..3
  unsigned long long xg_base = 0ull;
  unsigned lds_a0 = 0, lds_a1 = 0;
  if (w < 4) {
    xg_base = (unsigned long long)(size_t)(xg + (size_t)tg * 512 +
                                          (size_t)wgid * 16);
    lds_a0 = (unsigned)(size_t)&xgl[0][tg][0][0];
    lds_a1 = (unsigned)(size_t)&xgl[1][tg][0][0];
    // prologue: stage step 0 into buffer 0
    tdm_load_xg_tile(xg_base, lds_a0);
  }

  float* hT = out + (size_t)B_ * T_ * H_;

  for (int t = 0; t < T_; ++t) {
    const __bf16* hread = (t & 1) ? hbuf1 : hbuf0;
    __bf16* hwrite = (t & 1) ? hbuf0 : hbuf1;

    // ---- kick off TDM for step t+1 (overlaps the WMMA chain below) --------
    if (w < 4 && t + 1 < T_) {
      unsigned long long ga = xg_base +
          ((unsigned long long)(t + 1) * G4_) * sizeof(float);
      tdm_load_xg_tile(ga, ((t + 1) & 1) ? lds_a1 : lds_a0);
    }

    // ---- recurrent GEMM: 2 C tiles per wave, 16 k-steps -------------------
    v8f acc0 = {}, acc1 = {};
#pragma unroll 4
    for (int kt = 0; kt < KT_; ++kt) {
      v16bf a0 = *(const v16bf*)(hread + frag_off(mtb,     kt, lane));
      v16bf a1 = *(const v16bf*)(hread + frag_off(mtb + 1, kt, lane));
      acc0 = wmma_bf16(a0, bfrag[kt], acc0);
      acc1 = wmma_bf16(a1, bfrag[kt], acc1);
    }

    // ---- deposit recurrent contribution into LDS exchange -----------------
    const int ldcol = gate * 16 + nn;
#pragma unroll
    for (int r = 0; r < 8; ++r) {
      int b0r = mtb * 16 + hi * 8 + r;
      int b1r = (mtb + 1) * 16 + hi * 8 + r;
      gx[b0r * 64 + ldcol] = acc0[r];
      gx[b1r * 64 + ldcol] = acc1[r];
    }

    // ---- wait for this step's TDM tile (older of <=2 in flight) -----------
    if (w < 4) {
      if (t + 1 < T_) __builtin_amdgcn_s_wait_tensorcnt(1);
      else            __builtin_amdgcn_s_wait_tensorcnt(0);
    }
    __syncthreads();

    // ---- gates + cell update: 1024 (b,jj) elems, 4 per thread -------------
    const int xb = t & 1;
#pragma unroll
    for (int i = 0; i < 4; ++i) {
      int e = tid + i * 256;
      int b = e >> 4, jj = e & 15;
      float it = sigf(gx[b * 64 + jj]      + xgl[xb][0][b][jj]);
      float ft = sigf(gx[b * 64 + 16 + jj] + xgl[xb][1][b][jj]);
      float gt = tanhf(gx[b * 64 + 32 + jj] + xgl[xb][2][b][jj]);
      float ot = sigf(gx[b * 64 + 48 + jj] + xgl[xb][3][b][jj]);
      float cnew = ft * cst[e] + it * gt;
      float hnew = ot * tanhf(cnew);
      cst[e] = cnew;
      int j = wgid * 16 + jj;
      out[((size_t)b * T_ + t) * H_ + j] = ot;  // reference emits o_t per step
      if (t == T_ - 1) hT[b * H_ + j] = hnew;
      // scatter h_new into A-fragment layout for next step's GEMM
      int kk = j & 31, kt = j >> 5, grp = kk >> 3;
      int ln = (b & 15) + 16 * (grp & 1);
      int p = (kk & 7) + 8 * (grp >> 1);
      hwrite[((size_t)((b >> 4) * KT_ + kt) * 32 + ln) * 16 + p] = (__bf16)hnew;
    }

    // ---- device-wide step barrier (32 arrivals) ---------------------------
    __syncthreads();
    if (tid == 0) {
      __hip_atomic_fetch_add(counter, 1u, __ATOMIC_RELEASE,
                             __HIP_MEMORY_SCOPE_AGENT);
      unsigned tgt = (unsigned)(t + 1) * NWG_SCAN;
      while (__hip_atomic_load(counter, __ATOMIC_ACQUIRE,
                               __HIP_MEMORY_SCOPE_AGENT) < tgt) {
        __builtin_amdgcn_s_sleep(1);
      }
    }
    __syncthreads();
  }
}

// ---------------------------------------------------------------------------
extern "C" void kernel_launch(void* const* d_in, const int* in_sizes, int n_in,
                              void* d_out, int out_size, void* d_ws,
                              size_t ws_size, hipStream_t stream) {
  (void)in_sizes; (void)n_in; (void)out_size; (void)ws_size;
  const float* inputs = (const float*)d_in[0];  // (B,T,I)
  const float* w_i    = (const float*)d_in[1];  // (I,4H)
  const float* b_i    = (const float*)d_in[2];  // (4H)
  const float* w_h    = (const float*)d_in[3];  // (H,4H)
  const float* b_h    = (const float*)d_in[4];  // (4H)
  float* out = (float*)d_out;                   // (B,T,H) ++ (B,H)

  char* ws = (char*)d_ws;
  unsigned* counter = (unsigned*)ws;                       // 256 B slot
  __bf16* hbuf0 = (__bf16*)(ws + 256);                     // 64KiB
  __bf16* hbuf1 = hbuf0 + (size_t)B_ * H_;                 // 64KiB
  __bf16* whp = (__bf16*)(ws + 256 + 131072);              // 2 MiB
  __bf16* wip = whp + (size_t)I_ * G4_;                    // 2 MiB
  __bf16* ap  = wip + (size_t)I_ * G4_;                    // 128 MiB
  float*  xg  = (float*)((char*)(ap + (size_t)M_X * I_));  // 1 GiB

  // zero h state + barrier counter (must run every replay)
  lstm_init_kernel<<<64, 256, 0, stream>>>(counter, hbuf0, 2 * B_ * H_);

  // pack operands into WMMA fragment layouts (bf16)
  int a_threads = (M_X / 16) * KT_ * 32;                   // 4,194,304
  pack_a_kernel<<<a_threads / 256, 256, 0, stream>>>(inputs, ap, M_X / 16, I_);
  int w_threads = (G4_ / 16) * KT_ * 32;                   // 65,536
  pack_b_kernel<<<w_threads / 256, 256, 0, stream>>>(w_i, wip, G4_ / 16, G4_);
  pack_b_kernel<<<w_threads / 256, 256, 0, stream>>>(w_h, whp, G4_ / 16, G4_);

  // phase 1: big input GEMM (+ fused biases)
  gemm_xgates_kernel<<<2048 * 16, 256, 0, stream>>>(ap, wip, b_i, b_h, xg);

  // phase 2: persistent recurrent scan with TDM-staged x_gates
  lstm_scan_kernel<<<NWG_SCAN, 256, 0, stream>>>(xg, whp, hbuf0, hbuf1, out,
                                                 counter);
}
